// GAT_FP_37314675867864
// MI455X (gfx1250) — matrix operand order repaired and checked by reading, not compiled, not run on hardware
//
#include <hip/hip_runtime.h>
#include <hip/hip_bf16.h>

// ---------------------------------------------------------------------------
// GAT forward for MI455X (gfx1250, wave32, WMMA).
// Dense projections -> v_wmma_f32_16x16x32_bf16. All GEMM operands are padded
// in memory (rows to 64, K to 32 elements) so LDS staging is guard-free:
// b128 global load -> b128 LDS store -> ds_load_b128 fragments -> WMMA.
// Graph segment ops -> CSR built on device, block-per-node streaming kernels.
// ---------------------------------------------------------------------------

typedef __bf16 bf16;
typedef __attribute__((ext_vector_type(16))) __bf16 v16bf;
typedef __attribute__((ext_vector_type(8)))  float  v8f;
typedef __attribute__((ext_vector_type(4)))  unsigned int v4u;

union Frag { v16bf v; v4u q[2]; };

#define GIN 2024
#define KAP 1024      // padded pair-stride for IN-based GEMMs (2048 bf16)
#define GH 4
#define GF0 256
#define GF1 8
#define GOUT 6

// ---------------- utility kernels ----------------

__global__ void k_fill_f32(float* p, float v, long n) {
    long i = (long)blockIdx.x * blockDim.x + threadIdx.x;
    if (i < n) p[i] = v;
}

__global__ void k_fill_i32(int* p, int v, long n) {
    long i = (long)blockIdx.x * blockDim.x + threadIdx.x;
    if (i < n) p[i] = v;
}

__global__ void k_f2bf(const float* __restrict__ in, bf16* __restrict__ out, long n) {
    long i = (long)blockIdx.x * blockDim.x + threadIdx.x;
    if (i < n) out[i] = (bf16)in[i];
}

// Pack f32 weight W[K][Nc] (row-major) into bf16-pair layout with padded
// stride: Bpk[n*Kpad + p] = (bf16)W[2p][n] | (bf16)W[2p+1][n] << 16.
__global__ void k_packB(const float* __restrict__ W, unsigned int* __restrict__ Bpk,
                        int K, int Nc, int Kpad) {
    long Kp = K >> 1;
    long t = (long)blockIdx.x * blockDim.x + threadIdx.x;
    if (t >= (long)Nc * Kp) return;
    int n = (int)(t / Kp);
    int p = (int)(t % Kp);
    bf16 lo = (bf16)W[(long)(2 * p) * Nc + n];
    bf16 hi = (bf16)W[(long)(2 * p + 1) * Nc + n];
    unsigned int ulo = __builtin_bit_cast(unsigned short, lo);
    unsigned int uhi = __builtin_bit_cast(unsigned short, hi);
    Bpk[(long)n * Kpad + p] = ulo | (uhi << 16);
}

// ---------------- graph preprocessing ----------------

__global__ void k_deg(const int* __restrict__ src, const int* __restrict__ dst,
                      float* dout, float* din, int* cnt, int E) {
    int e = blockIdx.x * blockDim.x + threadIdx.x;
    if (e >= E) return;
    atomicAdd(&dout[src[e]], 1.0f);
    atomicAdd(&din[dst[e]], 1.0f);
    atomicAdd(&cnt[dst[e]], 1);
}

__global__ void k_rsqrtdeg(const float* dout, const float* din,
                           float* doutis, float* dinis, int N) {
    int i = blockIdx.x * blockDim.x + threadIdx.x;
    if (i >= N) return;
    doutis[i] = rsqrtf(fmaxf(dout[i], 1.0f));
    dinis[i]  = rsqrtf(fmaxf(din[i], 1.0f));
}

__global__ void k_scan(const int* __restrict__ cnt, int* __restrict__ off, int N) {
    if (blockIdx.x == 0 && threadIdx.x == 0) {
        int s = 0;
        for (int i = 0; i < N; ++i) { off[i] = s; s += cnt[i]; }
        off[N] = s;
    }
}

__global__ void k_csr(const int* __restrict__ dst, const int* __restrict__ off,
                      int* cursor, int* eid, int E) {
    int e = blockIdx.x * blockDim.x + threadIdx.x;
    if (e >= E) return;
    int d = dst[e];
    int p = atomicAdd(&cursor[d], 1);
    eid[off[d] + p] = e;
}

// GraphConv(norm=both) + 0.5*(x + agg) + L1 row-normalize + maskFilter, emit
// bf16 into a row-padded (2048 cols) buffer; pad columns written as zero.
// One block (256 threads) per node; 2048 cols -> 8 per thread.
__global__ __launch_bounds__(256)
void k_hmask(const float* __restrict__ x, const int* __restrict__ src,
             const int* __restrict__ off, const int* __restrict__ eid,
             const float* __restrict__ doutis, const float* __restrict__ dinis,
             const float* __restrict__ tm, const float* __restrict__ am,
             const float* __restrict__ vm, bf16* __restrict__ hmask, int N) {
    __shared__ float red[256];
    int n = blockIdx.x;
    int tid = threadIdx.x;
    int e0 = off[n], e1 = off[n + 1];
    float dr = dinis[n];
    float v[8];
    float pabs = 0.0f;
    #pragma unroll
    for (int j = 0; j < 8; ++j) {
        int f = tid + j * 256;
        float val = 0.0f;
        if (f < GIN) {
            float a = 0.0f;
            for (int ii = e0; ii < e1; ++ii) {
                int s = src[eid[ii]];
                a += x[(long)s * GIN + f] * doutis[s];
            }
            val = 0.5f * (x[(long)n * GIN + f] + a * dr);
            pabs += fabsf(val);
        }
        v[j] = val;
    }
    red[tid] = pabs;
    __syncthreads();
    for (int st = 128; st > 0; st >>= 1) {
        if (tid < st) red[tid] += red[tid + st];
        __syncthreads();
    }
    float inv = 1.0f / fmaxf(red[0], 1e-12f);
    #pragma unroll
    for (int j = 0; j < 8; ++j) {
        int f = tid + j * 256;
        float m = (f < GIN) ? (tm[f] + am[f] + vm[f]) : 0.0f;
        hmask[(long)n * (2 * KAP) + f] = (bf16)(v[j] * inv * m);
    }
}

// ---------------- WMMA GEMM ----------------
// C[M,Nc] = A @ W + bias. A given as bf16-pairs with padded row stride As_,
// B pre-packed as pairs with padded stride Bs_; all pads are zero so the
// K-loop needs no tail handling and staging needs no guards.
// Block: 128 threads = 4 waves in 2x2; wave tile 32x32; block tile 64x64.
#define TM 64
#define TN 64
#define KP 16          // pairs per k-step (K-step = 32 elements)
#define LDP 20         // padded LDS row stride in uints (16B-aligned)

__device__ inline void store_tile(float* C, const float* bias, v8f acc,
                                  int mbase, int nbase, int M, int Nc, int lane) {
    int nc = nbase + (lane & 15);
    if (nc >= Nc) return;
    float bv = bias[nc];
    #pragma unroll
    for (int i = 0; i < 8; ++i) {
        int mr = mbase + (lane >> 4) * 8 + i;
        if (mr < M) C[(long)mr * Nc + nc] = acc[i] + bv;
    }
}

__global__ __launch_bounds__(128)
void k_gemm_bf16(const unsigned int* __restrict__ Apk, int As_,
                 const unsigned int* __restrict__ Bpk, int Bs_,
                 const float* __restrict__ bias, float* __restrict__ C,
                 int M, int Nc, int Kpp) {
    __shared__ __align__(16) unsigned int As[TM][LDP];
    __shared__ __align__(16) unsigned int Bs[TN][LDP];
    int tid  = threadIdx.x;
    int lane = tid & 31;
    int wave = tid >> 5;              // 0..3
    int wm = (wave & 1) * 32;
    int wn = (wave >> 1) * 32;
    int m0 = blockIdx.x * TM;
    int n0 = blockIdx.y * TN;
    int half = lane >> 4;
    int l15  = lane & 15;

    v8f a00 = {0.f,0.f,0.f,0.f,0.f,0.f,0.f,0.f};
    v8f a01 = a00, a10 = a00, a11 = a00;

    for (int k0p = 0; k0p < Kpp; k0p += KP) {
        // prefetch next k-step (speculative; dropped if OOB)
        if (k0p + KP < Kpp) {
            __builtin_prefetch((const void*)&Apk[(long)(m0 + (tid >> 1)) * As_ + k0p + KP], 0, 1);
            __builtin_prefetch((const void*)&Bpk[(long)(n0 + (tid >> 1)) * Bs_ + k0p + KP], 0, 1);
        }
        // guard-free wide staging: 64 rows x 4 b128 chunks per matrix
        #pragma unroll
        for (int i = tid; i < TM * 4; i += 128) {   // 2 iterations
            int r = i >> 2, cq = (i & 3) * 4;
            *(v4u*)&As[r][cq] = *(const v4u*)&Apk[(long)(m0 + r) * As_ + k0p + cq];
            *(v4u*)&Bs[r][cq] = *(const v4u*)&Bpk[(long)(n0 + r) * Bs_ + k0p + cq];
        }
        __syncthreads();

        // Fragments: A lane pairs {half*4..+3, 8+half*4..+3}; B lane pairs {half*4..+3, 8+half*4..+3}
        Frag af0, af1, bq0, bq1;
        {
            int r0 = wm + l15, r1 = wm + 16 + l15;
            af0.q[0] = *(const v4u*)&As[r0][half * 4];
            af0.q[1] = *(const v4u*)&As[r0][8 + half * 4];
            af1.q[0] = *(const v4u*)&As[r1][half * 4];
            af1.q[1] = *(const v4u*)&As[r1][8 + half * 4];
            int c0 = wn + l15, c1 = wn + 16 + l15;
            bq0.q[0] = *(const v4u*)&Bs[c0][half * 4];
            bq0.q[1] = *(const v4u*)&Bs[c0][8 + half * 4];
            bq1.q[0] = *(const v4u*)&Bs[c1][half * 4];
            bq1.q[1] = *(const v4u*)&Bs[c1][8 + half * 4];
        }
        a00 = __builtin_amdgcn_wmma_f32_16x16x32_bf16(false, af0.v, false, bq0.v, (short)0, a00, false, false);
        a01 = __builtin_amdgcn_wmma_f32_16x16x32_bf16(false, af0.v, false, bq1.v, (short)0, a01, false, false);
        a10 = __builtin_amdgcn_wmma_f32_16x16x32_bf16(false, af1.v, false, bq0.v, (short)0, a10, false, false);
        a11 = __builtin_amdgcn_wmma_f32_16x16x32_bf16(false, af1.v, false, bq1.v, (short)0, a11, false, false);
        __syncthreads();
    }

    store_tile(C, bias, a00, m0 + wm,      n0 + wn,      M, Nc, lane);
    store_tile(C, bias, a01, m0 + wm,      n0 + wn + 16, M, Nc, lane);
    store_tile(C, bias, a10, m0 + wm + 16, n0 + wn,      M, Nc, lane);
    store_tile(C, bias, a11, m0 + wm + 16, n0 + wn + 16, M, Nc, lane);
}

// ---------------- GATv2 edge ops ----------------

__global__ void k_logits(const float* __restrict__ fs, const float* __restrict__ fd,
                         const int* __restrict__ src, const int* __restrict__ dst,
                         const float* __restrict__ attn, float* __restrict__ logits,
                         int E, int F) {
    int t = blockIdx.x * blockDim.x + threadIdx.x;
    if (t >= E * GH) return;
    int e = t / GH, h = t % GH;
    const float* ps = fs + (long)src[e] * GH * F + (long)h * F;
    const float* pd = fd + (long)dst[e] * GH * F + (long)h * F;
    const float* pa = attn + (long)h * F;
    float s = 0.0f;
    for (int f = 0; f < F; ++f) {
        float u = ps[f] + pd[f];
        u = (u > 0.0f) ? u : 0.2f * u;
        s += u * pa[f];
    }
    logits[(long)e * GH + h] = s;
}

__global__ void k_softmax(float* __restrict__ logits, const int* __restrict__ off,
                          const int* __restrict__ eid, int N) {
    int t = blockIdx.x * blockDim.x + threadIdx.x;
    if (t >= N * GH) return;
    int n = t / GH, h = t % GH;
    int e0 = off[n], e1 = off[n + 1];
    float mx = -3.4e38f;
    for (int ii = e0; ii < e1; ++ii)
        mx = fmaxf(mx, logits[(long)eid[ii] * GH + h]);
    float sum = 0.0f;
    for (int ii = e0; ii < e1; ++ii) {
        long idx = (long)eid[ii] * GH + h;
        float ex = expf(logits[idx] - mx);
        logits[idx] = ex;
        sum += ex;
    }
    float inv = 1.0f / sum;
    for (int ii = e0; ii < e1; ++ii)
        logits[(long)eid[ii] * GH + h] *= inv;
}

__global__ __launch_bounds__(256)
void k_aggr(const float* __restrict__ fs, const float* __restrict__ alpha,
            const int* __restrict__ src, const int* __restrict__ off,
            const int* __restrict__ eid, float* __restrict__ out,
            int F, int useRes) {
    int n = blockIdx.x;
    int HF = GH * F;
    int e0 = off[n], e1 = off[n + 1];
    for (int c = threadIdx.x; c < HF; c += blockDim.x) {
        int h = c / F;
        float a = 0.0f;
        for (int ii = e0; ii < e1; ++ii) {
            int e = eid[ii];
            a += alpha[(long)e * GH + h] * fs[(long)src[e] * HF + c];
        }
        float r = useRes ? out[(long)n * HF + c] : 0.0f;
        r += a;
        out[(long)n * HF + c] = (r > 0.0f) ? r : 0.0f;
    }
}

__global__ void k_final(const float* __restrict__ h3, const float* __restrict__ h1,
                        const float* __restrict__ W, const float* __restrict__ b,
                        float* __restrict__ out, int N) {
    int n = blockIdx.x * blockDim.x + threadIdx.x;
    if (n >= N) return;
    float acc[GOUT];
    #pragma unroll
    for (int j = 0; j < GOUT; ++j) acc[j] = b[j];
    for (int f = 0; f < 32; ++f) {
        float v = h3[(long)n * 32 + f];
        #pragma unroll
        for (int j = 0; j < GOUT; ++j) acc[j] += v * W[f * GOUT + j];
    }
    for (int f = 0; f < 32; ++f) {
        float v = h1[(long)n * 32 + f];
        #pragma unroll
        for (int j = 0; j < GOUT; ++j) acc[j] += v * W[(32 + f) * GOUT + j];
    }
    #pragma unroll
    for (int j = 0; j < GOUT; ++j) out[(long)n * GOUT + j] = acc[j];
}

// ---------------- host orchestration ----------------

static inline void launch_gemm(const unsigned int* Apk, int As_, const unsigned int* Bpk,
                               int Bs_, const float* bias, float* C,
                               int M, int Mpad, int Nc, int Ncpad, int Kpp, hipStream_t s) {
    dim3 grid(Mpad / TM, Ncpad / TN);
    k_gemm_bf16<<<grid, 128, 0, s>>>(Apk, As_, Bpk, Bs_, bias, C, M, Nc, Kpp);
}

extern "C" void kernel_launch(void* const* d_in, const int* in_sizes, int n_in,
                              void* d_out, int out_size, void* d_ws, size_t ws_size,
                              hipStream_t stream) {
    const int N = in_sizes[0] / GIN;
    const int E = in_sizes[1];
    const int HF0 = GH * GF0;          // 1024
    const int HF1 = GH * GF1;          // 32
    const int Mpad = ((N + 63) / 64) * 64;
    const int KP0 = HF0 / 2;           // 512 pairs (layer1 K, already x16-aligned)
    const int NC1PAD = 64;             // padded Nc for 32-wide GEMMs

    const float* x    = (const float*)d_in[0];
    const int*   src  = (const int*)d_in[1];
    const int*   dst  = (const int*)d_in[2];
    const float* tm   = (const float*)d_in[3];
    const float* am   = (const float*)d_in[4];
    const float* vm   = (const float*)d_in[5];
    const float* Wsrc0 = (const float*)d_in[6];  const float* bsrc0 = (const float*)d_in[7];
    const float* Wdst0 = (const float*)d_in[8];  const float* bdst0 = (const float*)d_in[9];
    const float* attn0 = (const float*)d_in[10];
    const float* Wres0 = (const float*)d_in[11]; const float* bres0 = (const float*)d_in[12];
    const float* Wsrc1 = (const float*)d_in[13]; const float* bsrc1 = (const float*)d_in[14];
    const float* Wdst1 = (const float*)d_in[15]; const float* bdst1 = (const float*)d_in[16];
    const float* attn1 = (const float*)d_in[17];
    const float* Wres1 = (const float*)d_in[18]; const float* bres1 = (const float*)d_in[19];
    const float* Wsrc2 = (const float*)d_in[20]; const float* bsrc2 = (const float*)d_in[21];
    const float* Wdst2 = (const float*)d_in[22]; const float* bdst2 = (const float*)d_in[23];
    const float* attn2 = (const float*)d_in[24];
    const float* Wlin  = (const float*)d_in[25]; const float* blin  = (const float*)d_in[26];

    // ---- carve workspace ----
    char* ws = (char*)d_ws;
    size_t off = 0;
    auto carve = [&](size_t bytes) -> void* {
        void* p = ws + off;
        off = (off + bytes + 255) & ~(size_t)255;
        return p;
    };
    float* dout_f  = (float*)carve((size_t)N * 4);
    float* din_f   = (float*)carve((size_t)N * 4);
    float* doutis  = (float*)carve((size_t)N * 4);
    float* dinis   = (float*)carve((size_t)N * 4);
    int*   cnt     = (int*)carve((size_t)N * 4);
    int*   offcsr  = (int*)carve((size_t)(N + 1) * 4);
    int*   cursor  = (int*)carve((size_t)N * 4);
    int*   eid     = (int*)carve((size_t)E * 4);
    float* logits  = (float*)carve((size_t)E * GH * 4);
    // padded A operands (as uint pair arrays)
    unsigned int* hmaskU = (unsigned int*)carve((size_t)Mpad * KAP * 4);
    unsigned int* h0bfU  = (unsigned int*)carve((size_t)Mpad * KP0 * 4);
    float* fs0     = (float*)carve((size_t)N * HF0 * 4);
    float* fd0     = (float*)carve((size_t)N * HF0 * 4);
    float* out0    = (float*)carve((size_t)N * HF0 * 4);   // res0 -> layer0 out (h0)
    float* fs2     = (float*)carve((size_t)N * HF1 * 4);
    float* fd2     = (float*)carve((size_t)N * HF1 * 4);
    float* h3      = (float*)carve((size_t)N * HF1 * 4);
    float* fs1     = (float*)carve((size_t)N * HF1 * 4);
    float* fd1     = (float*)carve((size_t)N * HF1 * 4);
    float* out1    = (float*)carve((size_t)N * HF1 * 4);   // res1 -> layer1 out
    // padded packed weights [Ncpad][Kpad]
    unsigned int* Wsrc0p = (unsigned int*)carve((size_t)HF0 * KAP * 4);
    unsigned int* Wdst0p = (unsigned int*)carve((size_t)HF0 * KAP * 4);
    unsigned int* Wres0p = (unsigned int*)carve((size_t)HF0 * KAP * 4);
    unsigned int* Wsrc1p = (unsigned int*)carve((size_t)NC1PAD * KP0 * 4);
    unsigned int* Wdst1p = (unsigned int*)carve((size_t)NC1PAD * KP0 * 4);
    unsigned int* Wres1p = (unsigned int*)carve((size_t)NC1PAD * KP0 * 4);
    unsigned int* Wsrc2p = (unsigned int*)carve((size_t)NC1PAD * KAP * 4);
    unsigned int* Wdst2p = (unsigned int*)carve((size_t)NC1PAD * KAP * 4);
    (void)ws_size; (void)n_in; (void)out_size;

    auto g1 = [](long n) { return dim3((unsigned)((n + 255) / 256)); };

    // ---- zero padded operand buffers (pads must stay zero) ----
    k_fill_i32<<<g1((long)Mpad * KAP), 256, 0, stream>>>((int*)hmaskU, 0, (long)Mpad * KAP);
    k_fill_i32<<<g1((long)Mpad * KP0), 256, 0, stream>>>((int*)h0bfU, 0, (long)Mpad * KP0);
    k_fill_i32<<<g1((long)HF0 * KAP), 256, 0, stream>>>((int*)Wsrc0p, 0, (long)HF0 * KAP);
    k_fill_i32<<<g1((long)HF0 * KAP), 256, 0, stream>>>((int*)Wdst0p, 0, (long)HF0 * KAP);
    k_fill_i32<<<g1((long)HF0 * KAP), 256, 0, stream>>>((int*)Wres0p, 0, (long)HF0 * KAP);
    k_fill_i32<<<g1((long)NC1PAD * KP0), 256, 0, stream>>>((int*)Wsrc1p, 0, (long)NC1PAD * KP0);
    k_fill_i32<<<g1((long)NC1PAD * KP0), 256, 0, stream>>>((int*)Wdst1p, 0, (long)NC1PAD * KP0);
    k_fill_i32<<<g1((long)NC1PAD * KP0), 256, 0, stream>>>((int*)Wres1p, 0, (long)NC1PAD * KP0);
    k_fill_i32<<<g1((long)NC1PAD * KAP), 256, 0, stream>>>((int*)Wsrc2p, 0, (long)NC1PAD * KAP);
    k_fill_i32<<<g1((long)NC1PAD * KAP), 256, 0, stream>>>((int*)Wdst2p, 0, (long)NC1PAD * KAP);

    // ---- graph preprocessing ----
    k_fill_f32<<<g1(N), 256, 0, stream>>>(dout_f, 0.0f, N);
    k_fill_f32<<<g1(N), 256, 0, stream>>>(din_f, 0.0f, N);
    k_fill_i32<<<g1(N), 256, 0, stream>>>(cnt, 0, N);
    k_fill_i32<<<g1(N), 256, 0, stream>>>(cursor, 0, N);
    k_deg<<<g1(E), 256, 0, stream>>>(src, dst, dout_f, din_f, cnt, E);
    k_rsqrtdeg<<<g1(N), 256, 0, stream>>>(dout_f, din_f, doutis, dinis, N);
    k_scan<<<1, 64, 0, stream>>>(cnt, offcsr, N);
    k_csr<<<g1(E), 256, 0, stream>>>(dst, offcsr, cursor, eid, E);

    // ---- weight packing to padded bf16-pair layout ----
    k_packB<<<g1((long)GIN / 2 * HF0), 256, 0, stream>>>(Wsrc0, Wsrc0p, GIN, HF0, KAP);
    k_packB<<<g1((long)GIN / 2 * HF0), 256, 0, stream>>>(Wdst0, Wdst0p, GIN, HF0, KAP);
    k_packB<<<g1((long)GIN / 2 * HF0), 256, 0, stream>>>(Wres0, Wres0p, GIN, HF0, KAP);
    k_packB<<<g1((long)HF0 / 2 * HF1), 256, 0, stream>>>(Wsrc1, Wsrc1p, HF0, HF1, KP0);
    k_packB<<<g1((long)HF0 / 2 * HF1), 256, 0, stream>>>(Wdst1, Wdst1p, HF0, HF1, KP0);
    k_packB<<<g1((long)HF0 / 2 * HF1), 256, 0, stream>>>(Wres1, Wres1p, HF0, HF1, KP0);
    k_packB<<<g1((long)GIN / 2 * HF1), 256, 0, stream>>>(Wsrc2, Wsrc2p, GIN, HF1, KAP);
    k_packB<<<g1((long)GIN / 2 * HF1), 256, 0, stream>>>(Wdst2, Wdst2p, GIN, HF1, KAP);

    // ---- GraphConv + L1 + maskFilter -> padded bf16 activations ----
    k_hmask<<<N, 256, 0, stream>>>(x, src, offcsr, eid, doutis, dinis,
                                   tm, am, vm, (bf16*)hmaskU, N);

    // ---- gat2 (cross-modal, no residual) -> h3 ----
    launch_gemm(hmaskU, KAP, Wsrc2p, KAP, bsrc2, fs2, N, Mpad, HF1, NC1PAD, KAP, stream);
    launch_gemm(hmaskU, KAP, Wdst2p, KAP, bdst2, fd2, N, Mpad, HF1, NC1PAD, KAP, stream);
    k_logits<<<g1((long)E * GH), 256, 0, stream>>>(fs2, fd2, src, dst, attn2, logits, E, GF1);
    k_softmax<<<g1((long)N * GH), 256, 0, stream>>>(logits, offcsr, eid, N);
    k_aggr<<<N, 256, 0, stream>>>(fs2, logits, src, offcsr, eid, h3, GF1, 0);

    // ---- layer0 (2024 -> 4x256, residual) -> out0 ----
    launch_gemm(hmaskU, KAP, Wsrc0p, KAP, bsrc0, fs0, N, Mpad, HF0, HF0, KAP, stream);
    launch_gemm(hmaskU, KAP, Wdst0p, KAP, bdst0, fd0, N, Mpad, HF0, HF0, KAP, stream);
    launch_gemm(hmaskU, KAP, Wres0p, KAP, bres0, out0, N, Mpad, HF0, HF0, KAP, stream);
    k_logits<<<g1((long)E * GH), 256, 0, stream>>>(fs0, fd0, src, dst, attn0, logits, E, GF0);
    k_softmax<<<g1((long)N * GH), 256, 0, stream>>>(logits, offcsr, eid, N);
    k_aggr<<<N, 256, 0, stream>>>(fs0, logits, src, offcsr, eid, out0, GF0, 1);

    // ---- layer1 (1024 -> 4x8, residual) -> out1 ----
    k_f2bf<<<g1((long)N * HF0), 256, 0, stream>>>(out0, (bf16*)h0bfU, (long)N * HF0);
    launch_gemm(h0bfU, KP0, Wsrc1p, KP0, bsrc1, fs1, N, Mpad, HF1, NC1PAD, KP0, stream);
    launch_gemm(h0bfU, KP0, Wdst1p, KP0, bdst1, fd1, N, Mpad, HF1, NC1PAD, KP0, stream);
    launch_gemm(h0bfU, KP0, Wres1p, KP0, bres1, out1, N, Mpad, HF1, NC1PAD, KP0, stream);
    k_logits<<<g1((long)E * GH), 256, 0, stream>>>(fs1, fd1, src, dst, attn1, logits, E, GF1);
    k_softmax<<<g1((long)N * GH), 256, 0, stream>>>(logits, offcsr, eid, N);
    k_aggr<<<N, 256, 0, stream>>>(fs1, logits, src, offcsr, eid, out1, GF1, 1);

    // ---- final linear ----
    k_final<<<g1(N), 256, 0, stream>>>(h3, out1, Wlin, blin, (float*)d_out, N);
}
